// CQAttention_73564199846305
// MI455X (gfx1250) — compile-verified, matrix operand order checked by reading, not run on hardware
//
#include <hip/hip_runtime.h>

// ---------------------------------------------------------------------------
// CQAttention for MI455X (gfx1250, wave32, WMMA).
// B=32, d=128, Lc=2048, Lq=512. All GEMMs on v_wmma_f32_16x16x32_bf16.
// All WMMA operands are stored K-contiguous so one fragment = 2x b128 loads.
// S (134 MB fp32) is materialized once and stays resident in the 192 MB L2.
// ---------------------------------------------------------------------------

#define BATCH 32
#define DDIM  128
#define LCTX  2048
#define LQRY  512
#define NEGF  (-1e30f)

typedef __attribute__((ext_vector_type(16))) __bf16          v16bf;
typedef __attribute__((ext_vector_type(16))) unsigned short  v16us;
typedef __attribute__((ext_vector_type(8)))  unsigned short  us8;
typedef __attribute__((ext_vector_type(8)))  float           v8f;

__device__ __forceinline__ unsigned short f32_to_bf16(float f) {
  unsigned int u = __float_as_uint(f);
  unsigned int r = u + 0x7fffu + ((u >> 16) & 1u);   // round-to-nearest-even
  return (unsigned short)(r >> 16);
}

// Fragment load per ISA 7.12.2 16-bit layouts: K-contiguous row of 32
// elements at p; lane's K-half selects +0 or +8; halves at +0 and +16.
// One fragment = two 16B loads.
__device__ __forceinline__ v16bf frag_ld(const unsigned short* p, int khalf) {
  us8 lo = *(const us8*)(p + khalf);
  us8 hi = *(const us8*)(p + 16 + khalf);
  v16us us = __builtin_shufflevector(lo, hi, 0, 1, 2, 3, 4, 5, 6, 7, 8, 9, 10,
                                     11, 12, 13, 14, 15);
  return __builtin_bit_cast(v16bf, us);
}

__device__ __forceinline__ v8f wmma_bf16(v16bf a, v16bf b, v8f c) {
  return __builtin_amdgcn_wmma_f32_16x16x32_bf16(false, a, false, b, (short)0,
                                                 c, false, false);
}

__device__ __forceinline__ float4 mul4(float4 a, float4 b) {
  return {a.x * b.x, a.y * b.y, a.z * b.z, a.w * b.w};
}

// ---------------- K0a: CwT[b][c][d] = bf16(C*w_mul), fused s0 --------------
__global__ void cqa_convT_c(const float* __restrict__ C,
                            const float* __restrict__ w_mul,
                            const float* __restrict__ w_c,
                            unsigned short* __restrict__ CwT,
                            float* __restrict__ s0) {
  int i = blockIdx.x * blockDim.x + threadIdx.x;   // i = b*LCTX + c
  if (i >= BATCH * LCTX) return;
  int b = i / LCTX, c = i % LCTX;
  float acc = 0.f;
  unsigned short* dst = CwT + (size_t)i * DDIM;
#pragma unroll
  for (int j = 0; j < DDIM / 8; ++j) {
    us8 v;
#pragma unroll
    for (int k = 0; k < 8; ++k) {
      int d = j * 8 + k;
      float f = C[((size_t)(b * DDIM + d)) * LCTX + c];
      v[k] = f32_to_bf16(f * w_mul[d]);
      acc += f * w_c[d];
    }
    *(us8*)(dst + j * 8) = v;
  }
  s0[i] = acc;
}

// ---------------- K0b: QbT[b][q][d] = bf16(Q), fused s1 --------------------
__global__ void cqa_convT_q(const float* __restrict__ Q,
                            const float* __restrict__ w_q,
                            unsigned short* __restrict__ QbT,
                            float* __restrict__ s1) {
  int i = blockIdx.x * blockDim.x + threadIdx.x;   // i = b*LQRY + q
  if (i >= BATCH * LQRY) return;
  int b = i / LQRY, q = i % LQRY;
  float acc = 0.f;
  unsigned short* dst = QbT + (size_t)i * DDIM;
#pragma unroll
  for (int j = 0; j < DDIM / 8; ++j) {
    us8 v;
#pragma unroll
    for (int k = 0; k < 8; ++k) {
      int d = j * 8 + k;
      float f = Q[((size_t)(b * DDIM + d)) * LQRY + q];
      v[k] = f32_to_bf16(f);
      acc += f * w_q[d];
    }
    *(us8*)(dst + j * 8) = v;
  }
  s1[i] = acc;
}

// ---------------- K0c: flat fp32 -> bf16 conversion (vectorized) -----------
__global__ void cqa_conv_flat(const float* __restrict__ src,
                              unsigned short* __restrict__ dst, size_t n8) {
  size_t i = (size_t)blockIdx.x * blockDim.x + threadIdx.x;
  if (i >= n8) return;
  const float4* s = (const float4*)(src + i * 8);
  float4 x0 = s[0], x1 = s[1];
  us8 v;
  v[0] = f32_to_bf16(x0.x); v[1] = f32_to_bf16(x0.y);
  v[2] = f32_to_bf16(x0.z); v[3] = f32_to_bf16(x0.w);
  v[4] = f32_to_bf16(x1.x); v[5] = f32_to_bf16(x1.y);
  v[6] = f32_to_bf16(x1.z); v[7] = f32_to_bf16(x1.w);
  *(us8*)(dst + i * 8) = v;
}

// ------------------- K1: S = (Cw)^T Q + s0 + s1 + bias ---------------------
// block = 128 threads (4 waves); block tile 16c x 64q; k-loop d in steps of 32.
__global__ void cqa_s_mat(const unsigned short* __restrict__ CwT,
                          const unsigned short* __restrict__ QbT,
                          const float* __restrict__ s0,
                          const float* __restrict__ s1,
                          const float* __restrict__ bias,
                          float* __restrict__ S) {
  __shared__ __align__(16) unsigned short lds_a[16 * 32];  // [c][d32]
  __shared__ __align__(16) unsigned short lds_b[64 * 32];  // [q][d32]
  int tid = threadIdx.x, lane = tid & 31, wave = tid >> 5;
  int qb64 = blockIdx.x * 64;
  int c0 = blockIdx.y * 16;
  int b = blockIdx.z;
  int m = lane & 15, khalf = (lane & 16) ? 8 : 0;
  v8f acc = {};
  for (int k0 = 0; k0 < DDIM; k0 += 32) {
    if (tid < 64) {
      int cl = tid >> 2, ch = tid & 3;
      *(us8*)(lds_a + cl * 32 + ch * 8) =
          *(const us8*)(CwT + ((size_t)(b * LCTX + c0 + cl)) * DDIM + k0 +
                        ch * 8);
    }
#pragma unroll
    for (int j = 0; j < 2; ++j) {
      int idx = tid + 128 * j;
      int ql = idx >> 2, ch = idx & 3;
      *(us8*)(lds_b + ql * 32 + ch * 8) =
          *(const us8*)(QbT + ((size_t)(b * LQRY + qb64 + ql)) * DDIM + k0 +
                        ch * 8);
    }
    __syncthreads();
    v16bf a  = frag_ld(lds_a + m * 32, khalf);                  // A(m=c,k=d)
    v16bf bb = frag_ld(lds_b + (wave * 16 + m) * 32, khalf);    // B(k=d,n=q)
    acc = wmma_bf16(a, bb, acc);
    __syncthreads();
  }
  int q0 = qb64 + wave * 16;
  int hi = (lane >> 4) & 1;
  int q = q0 + m;
  float s1v = s1[b * LQRY + q] + bias[0];
#pragma unroll
  for (int v = 0; v < 8; ++v) {
    int c = c0 + v + hi * 8;
    S[((size_t)(b * LCTX + c)) * LQRY + q] = acc[v] + s0[b * LCTX + c] + s1v;
  }
}

// ------------------- K2: per-row (c) softmax stats over q ------------------
__global__ void cqa_rowstats(const float* __restrict__ S,
                             const int* __restrict__ Qmask,
                             float* __restrict__ rowmax,
                             float* __restrict__ rowinv) {
  int tid = threadIdx.x, lane = tid & 31, wave = tid >> 5;
  int r = blockIdx.x * 8 + wave;            // r = b*LCTX + c
  int b = r / LCTX;
  const float* row = S + (size_t)r * LQRY;
  float m = -3.4e38f, s = 0.f;
#pragma unroll
  for (int it = 0; it < LQRY / 128; ++it) {
    int q = it * 128 + lane * 4;
    float4 x4 = *(const float4*)(row + q);
    float xv[4] = {x4.x, x4.y, x4.z, x4.w};
#pragma unroll
    for (int j = 0; j < 4; ++j) {
      float x = xv[j] + (Qmask[b * LQRY + q + j] ? 0.f : NEGF);
      float nm = fmaxf(m, x);
      s = s * __expf(m - nm) + __expf(x - nm);
      m = nm;
    }
  }
#pragma unroll
  for (int off = 16; off; off >>= 1) {
    float om = __shfl_xor(m, off, 32);
    float os = __shfl_xor(s, off, 32);
    float nm = fmaxf(m, om);
    s = s * __expf(m - nm) + os * __expf(om - nm);
    m = nm;
  }
  if (lane == 0) { rowmax[r] = m; rowinv[r] = 1.0f / s; }
}

// ------------------- K3: per-col (q) softmax stats over c ------------------
__global__ void cqa_colstats(const float* __restrict__ S,
                             const int* __restrict__ Cmask,
                             float* __restrict__ colmax,
                             float* __restrict__ colinv) {
  __shared__ float sm[256], ss[256];
  int tid = threadIdx.x;
  int ql = tid & 63, cp = tid >> 6;
  int q = blockIdx.x * 64 + ql;
  int b = blockIdx.y;
  float m = -3.4e38f, s = 0.f;
  for (int c = cp; c < LCTX; c += 4) {
    float x = S[((size_t)(b * LCTX + c)) * LQRY + q] +
              (Cmask[b * LCTX + c] ? 0.f : NEGF);
    float nm = fmaxf(m, x);
    s = s * __expf(m - nm) + __expf(x - nm);
    m = nm;
  }
  sm[tid] = m; ss[tid] = s;
  __syncthreads();
  if (cp == 0) {
#pragma unroll
    for (int j = 1; j < 4; ++j) {
      float om = sm[ql + 64 * j], os = ss[ql + 64 * j];
      float nm = fmaxf(m, om);
      s = s * __expf(m - nm) + os * __expf(om - nm);
      m = nm;
    }
    colmax[b * LQRY + q] = m;
    colinv[b * LQRY + q] = 1.0f / s;
  }
}

// ---- K4: T[q,d] = sum_c S2[c,q]*Ct[c,d]; store bf16 as Tb[b][d][q] --------
// block = 256 (8 waves); wave w owns d-slice [16w,16w+16).
__global__ void cqa_t_mat(const float* __restrict__ S,
                          const unsigned short* __restrict__ Cb,
                          const int* __restrict__ Cmask,
                          const float* __restrict__ colmax,
                          const float* __restrict__ colinv,
                          unsigned short* __restrict__ Tb) {
  __shared__ __align__(16) unsigned short lds_s2[16 * 32];  // [q][c32]
  int tid = threadIdx.x, lane = tid & 31, wave = tid >> 5;
  int q0 = blockIdx.x * 16;
  int b = blockIdx.y;
  int dd0 = wave * 16;
  int n = lane & 15, khalf = (lane & 16) ? 8 : 0, hi = (lane >> 4) & 1;
  float cmx = colmax[b * LQRY + q0 + (tid & 15)];
  float civ = colinv[b * LQRY + q0 + (tid & 15)];
  v8f acc = {};
  for (int c0 = 0; c0 < LCTX; c0 += 32) {
    {  // stage exp(S2) tile, K(c)-contiguous, packed 32-bit LDS stores
      int ql = tid & 15, cpair = tid >> 4;   // 16 q x 16 c-pairs
      int c = c0 + cpair * 2;
      int q = q0 + ql;
      float x0 = S[((size_t)(b * LCTX + c)) * LQRY + q] +
                 (Cmask[b * LCTX + c] ? 0.f : NEGF);
      float x1 = S[((size_t)(b * LCTX + c + 1)) * LQRY + q] +
                 (Cmask[b * LCTX + c + 1] ? 0.f : NEGF);
      unsigned int pk = (unsigned int)f32_to_bf16(__expf(x0 - cmx) * civ) |
                        ((unsigned int)f32_to_bf16(__expf(x1 - cmx) * civ)
                         << 16);
      *(unsigned int*)(lds_s2 + ql * 32 + cpair * 2) = pk;
    }
    __syncthreads();
    v16bf a = frag_ld(lds_s2 + n * 32, khalf);                 // A(m=q,k=c)
    const unsigned short* bp =
        Cb + ((size_t)(b * DDIM + dd0 + n)) * LCTX + c0;       // B(k=c,n=d)
    v16bf bb = frag_ld(bp, khalf);
    if (c0 + 32 < LCTX) __builtin_prefetch(bp + 32, 0, 1);
    acc = wmma_bf16(a, bb, acc);
    __syncthreads();
  }
  us8 tv;
#pragma unroll
  for (int v = 0; v < 8; ++v) tv[v] = f32_to_bf16(acc[v]);
  *(us8*)(Tb + ((size_t)(b * DDIM + dd0 + n)) * LQRY + q0 + 8 * hi) = tv;
}

// -------- K5: A = S1 Q^T, Bm = S1 T, emit [Ct, A, Ct*A, Ct*Bm] -------------
// block = 256 (8 waves); wave w owns d-slice [16w,16w+16).
__global__ void cqa_out(const float* __restrict__ S,
                        const float* __restrict__ C,
                        const unsigned short* __restrict__ Qb,
                        const unsigned short* __restrict__ Tb,
                        const int* __restrict__ Qmask,
                        const float* __restrict__ rowmax,
                        const float* __restrict__ rowinv,
                        float* __restrict__ out) {
  __shared__ __align__(16) unsigned short lds_s1[16 * 32];  // [c][q32]
  int tid = threadIdx.x, lane = tid & 31, wave = tid >> 5;
  int c0 = blockIdx.x * 16;
  int b = blockIdx.y;
  int dd0 = wave * 16;
  int n = lane & 15, khalf = (lane & 16) ? 8 : 0, hi = (lane >> 4) & 1;
  int cl_st = tid >> 4;                     // staging row (c) for this thread
  float rmx = rowmax[b * LCTX + c0 + cl_st];
  float riv = rowinv[b * LCTX + c0 + cl_st];
  v8f accA = {}, accB = {};
  for (int q0 = 0; q0 < LQRY; q0 += 32) {
    {  // stage exp(S1) tile, K(q)-contiguous
      int qpair = tid & 15;
      int q = q0 + qpair * 2;
      const float* sp = S + ((size_t)(b * LCTX + c0 + cl_st)) * LQRY + q;
      float2 x2 = *(const float2*)sp;
      if (q0 + 32 < LQRY) __builtin_prefetch(sp + 32, 0, 1);
      float x0 = x2.x + (Qmask[b * LQRY + q] ? 0.f : NEGF);
      float x1 = x2.y + (Qmask[b * LQRY + q + 1] ? 0.f : NEGF);
      unsigned int pk = (unsigned int)f32_to_bf16(__expf(x0 - rmx) * riv) |
                        ((unsigned int)f32_to_bf16(__expf(x1 - rmx) * riv)
                         << 16);
      *(unsigned int*)(lds_s1 + cl_st * 32 + qpair * 2) = pk;
    }
    __syncthreads();
    v16bf a  = frag_ld(lds_s1 + n * 32, khalf);                    // A(m=c,k=q)
    v16bf bq = frag_ld(Qb + ((size_t)(b * DDIM + dd0 + n)) * LQRY + q0,
                       khalf);                                     // B(k=q,n=d)
    v16bf bt = frag_ld(Tb + ((size_t)(b * DDIM + dd0 + n)) * LQRY + q0,
                       khalf);                                     // B(k=q,n=d)
    accA = wmma_bf16(a, bq, accA);
    accB = wmma_bf16(a, bt, accB);
    __syncthreads();
  }
  int dd = dd0 + n;
  int cb = c0 + 8 * hi;
  const float* cp = C + ((size_t)(b * DDIM + dd)) * LCTX + cb;
  float4 ct0 = *(const float4*)cp;
  float4 ct1 = *(const float4*)(cp + 4);
  float4 a0 = {accA[0], accA[1], accA[2], accA[3]};
  float4 a1 = {accA[4], accA[5], accA[6], accA[7]};
  float4 b0 = {accB[0], accB[1], accB[2], accB[3]};
  float4 b1 = {accB[4], accB[5], accB[6], accB[7]};
  size_t ob = (size_t)b * 4 * DDIM * LCTX;
  float* o0 = out + ob + (size_t)dd * LCTX + cb;
  float* o1 = out + ob + (size_t)(DDIM + dd) * LCTX + cb;
  float* o2 = out + ob + (size_t)(2 * DDIM + dd) * LCTX + cb;
  float* o3 = out + ob + (size_t)(3 * DDIM + dd) * LCTX + cb;
  *(float4*)o0 = ct0;            *(float4*)(o0 + 4) = ct1;
  *(float4*)o1 = a0;             *(float4*)(o1 + 4) = a1;
  *(float4*)o2 = mul4(ct0, a0);  *(float4*)(o2 + 4) = mul4(ct1, a1);
  *(float4*)o3 = mul4(ct0, b0);  *(float4*)(o3 + 4) = mul4(ct1, b1);
}

// ---------------------------------------------------------------------------

extern "C" void kernel_launch(void* const* d_in, const int* in_sizes, int n_in,
                              void* d_out, int out_size, void* d_ws,
                              size_t ws_size, hipStream_t stream) {
  (void)in_sizes; (void)n_in; (void)out_size; (void)ws_size;
  const float* C     = (const float*)d_in[0];
  const float* Q     = (const float*)d_in[1];
  const int*   Cmask = (const int*)d_in[2];
  const int*   Qmask = (const int*)d_in[3];
  const float* w_c   = (const float*)d_in[4];
  const float* w_q   = (const float*)d_in[5];
  const float* w_mul = (const float*)d_in[6];
  const float* bias  = (const float*)d_in[7];
  float* out = (float*)d_out;

  char* ws = (char*)d_ws;
  size_t off = 0;
  auto take = [&](size_t bytes) -> void* {
    void* p = ws + off;
    off += (bytes + 255) & ~(size_t)255;
    return p;
  };
  const size_t nBDC = (size_t)BATCH * DDIM * LCTX;
  const size_t nBDQ = (size_t)BATCH * DDIM * LQRY;
  const size_t nS   = (size_t)BATCH * LCTX * LQRY;

  unsigned short* CwT = (unsigned short*)take(nBDC * 2);  // [b][c][d]
  unsigned short* Cb  = (unsigned short*)take(nBDC * 2);  // [b][d][c]
  unsigned short* QbT = (unsigned short*)take(nBDQ * 2);  // [b][q][d]
  unsigned short* Qb  = (unsigned short*)take(nBDQ * 2);  // [b][d][q]
  float* S      = (float*)take(nS * 4);
  float* s0     = (float*)take((size_t)BATCH * LCTX * 4);
  float* s1     = (float*)take((size_t)BATCH * LQRY * 4);
  float* rowmax = (float*)take((size_t)BATCH * LCTX * 4);
  float* rowinv = (float*)take((size_t)BATCH * LCTX * 4);
  float* colmax = (float*)take((size_t)BATCH * LQRY * 4);
  float* colinv = (float*)take((size_t)BATCH * LQRY * 4);
  unsigned short* Tb = (unsigned short*)take(nBDQ * 2);   // [b][d][q]

  cqa_convT_c<<<BATCH * LCTX / 256, 256, 0, stream>>>(C, w_mul, w_c, CwT, s0);
  cqa_convT_q<<<BATCH * LQRY / 256, 256, 0, stream>>>(Q, w_q, QbT, s1);
  cqa_conv_flat<<<(unsigned)(nBDC / 8 / 256), 256, 0, stream>>>(C, Cb,
                                                                nBDC / 8);
  cqa_conv_flat<<<(unsigned)(nBDQ / 8 / 256), 256, 0, stream>>>(Q, Qb,
                                                                nBDQ / 8);

  cqa_s_mat<<<dim3(LQRY / 64, LCTX / 16, BATCH), 128, 0, stream>>>(
      CwT, QbT, s0, s1, bias, S);

  cqa_rowstats<<<BATCH * LCTX / 8, 256, 0, stream>>>(S, Qmask, rowmax, rowinv);
  cqa_colstats<<<dim3(LQRY / 64, BATCH), 256, 0, stream>>>(S, Cmask, colmax,
                                                           colinv);

  cqa_t_mat<<<dim3(LQRY / 16, BATCH), 256, 0, stream>>>(S, Cb, Cmask, colmax,
                                                        colinv, Tb);

  cqa_out<<<dim3(LCTX / 16, BATCH), 256, 0, stream>>>(S, C, Qb, Tb, Qmask,
                                                      rowmax, rowinv, out);
}
